// SelfAttention_19284403159196
// MI455X (gfx1250) — compile-verified, matrix operand order checked by reading, not run on hardware
//
#include <hip/hip_runtime.h>

typedef __bf16 bf16;
typedef __attribute__((ext_vector_type(16))) __bf16 v16bf;
typedef __attribute__((ext_vector_type(8)))  __bf16 v8bf;
typedef __attribute__((ext_vector_type(8)))  float  v8f;
typedef int v4i __attribute__((vector_size(4 * sizeof(int))));

constexpr int Bn = 4, Tn = 2048, Cn = 1024, Hn = 16, HDn = 64;
constexpr int N3C   = 3 * Cn;     // 3072
constexpr int Mrows = Bn * Tn;    // 8192
constexpr int KP    = 40;         // padded LDS row stride (80B, 16B-aligned rows)

// ---------------------------------------------------------------------------
// CDNA5 async global->LDS copy (ASYNCcnt-tracked), with safe fallback.
// Parameter types per hipcc diagnostic: generic `int4*` pointers.
// ---------------------------------------------------------------------------
#if defined(__has_builtin)
#if __has_builtin(__builtin_amdgcn_global_load_async_to_lds_b128) && \
    __has_builtin(__builtin_amdgcn_s_wait_asynccnt)
#define HAVE_ASYNC_LDS 1
#endif
#endif
#ifndef HAVE_ASYNC_LDS
#define HAVE_ASYNC_LDS 0
#endif

__device__ __forceinline__ void async_copy16(bf16* lds, const bf16* g) {
#if HAVE_ASYNC_LDS
  __builtin_amdgcn_global_load_async_to_lds_b128((v4i*)g, (v4i*)lds, 0, 0);
#else
  *(v8bf*)lds = *(const v8bf*)g;
#endif
}

__device__ __forceinline__ void wait_async_all() {
#if HAVE_ASYNC_LDS
  __builtin_amdgcn_s_wait_asynccnt(0);
#endif
}

// ---------------------------------------------------------------------------
// WMMA helper: D = A(16x32 bf16) * B(32x16 bf16) + C(16x16 f32)
// ---------------------------------------------------------------------------
__device__ __forceinline__ v8f wmma_bf16(v16bf a, v16bf b, v8f c) {
  return __builtin_amdgcn_wmma_f32_16x16x32_bf16(false, a, false, b, (short)0, c,
                                                 false, false);
}

// 16x32 bf16 A-fragment from a row-major matrix (global or LDS). ISA layout:
// lanes 0-15: row m=lane, K chunks {0..7,16..23}; lanes 16-31: row m=lane-16,
// K chunks {8..15,24..31}. Two 16-byte loads per lane.
__device__ __forceinline__ v16bf load_frag(const bf16* __restrict__ base, int ld) {
  int lane = threadIdx.x & 31;
  int m  = lane & 15;
  int kb = (lane >> 4) << 3;               // 0 or 8
  const bf16* p = base + (size_t)m * ld + kb;
  v8bf lo = *(const v8bf*)(p);
  v8bf hi = *(const v8bf*)(p + 16);
  v16bf r;
#pragma unroll
  for (int i = 0; i < 8; ++i) { r[i] = lo[i]; r[i + 8] = hi[i]; }
  return r;
}

// ---------------------------------------------------------------------------
// f32 -> bf16 elementwise convert
// ---------------------------------------------------------------------------
__global__ void cvt_f32_bf16(const float* __restrict__ in, bf16* __restrict__ out,
                             int n) {
  int i = blockIdx.x * blockDim.x + threadIdx.x;
  if (i < n) out[i] = (bf16)in[i];
}

// in: [rows, cols] f32 row-major  ->  out: [cols, rows] bf16 row-major
__global__ void transpose_cvt(const float* __restrict__ in, bf16* __restrict__ out,
                              int rows, int cols) {
  int i = blockIdx.x * blockDim.x + threadIdx.x;
  if (i >= rows * cols) return;
  int n = i / rows;            // original column
  int k = i - n * rows;        // original row
  out[i] = (bf16)in[(size_t)k * cols + n];
}

// ---------------------------------------------------------------------------
// Shared GEMM core: block = 256 threads computes a 128(M) x 64(N) tile of
// A[M,K] @ B[K,N] with B supplied transposed (BT[N,K]).  K-loop staged through
// double-buffered LDS via async copies; wave w owns rows [w*16, w*16+16) and
// accumulates 4 16x16 f32 tiles covering all 64 N columns.
// ---------------------------------------------------------------------------
__device__ __forceinline__ void stage_tile(bf16* sa, bf16* sb,
                                           const bf16* __restrict__ Ag,
                                           const bf16* __restrict__ Bg, int tid) {
  // A tile: 128 rows x 32 cols -> 512 16B chunks (2 per thread)
#pragma unroll
  for (int c = tid; c < 512; c += 256) {
    int row = c >> 2, col = (c & 3) * 8;
    async_copy16(sa + row * KP + col, Ag + (size_t)row * Cn + col);
  }
  // B tile: 64 rows x 32 cols -> 256 16B chunks (1 per thread)
  {
    int row = tid >> 2, col = (tid & 3) * 8;
    async_copy16(sb + row * KP + col, Bg + (size_t)row * Cn + col);
  }
}

__device__ __forceinline__ void gemm_core(const bf16* __restrict__ Ag,
                                          const bf16* __restrict__ Bg,
                                          v8f* acc) {
  __shared__ bf16 sA[2][128 * KP];   // 2 x 10 KB
  __shared__ bf16 sB[2][64 * KP];    // 2 x  5 KB
  int tid = threadIdx.x;
  int wib = tid >> 5;

  stage_tile(sA[0], sB[0], Ag, Bg, tid);
  wait_async_all();
  __syncthreads();

  for (int k0 = 0; k0 < Cn; k0 += 32) {
    int cur = (k0 >> 5) & 1;
    int nxt = cur ^ 1;
    if (k0 + 32 < Cn)   // prefetch next k-slab while computing this one
      stage_tile(sA[nxt], sB[nxt], Ag + k0 + 32, Bg + k0 + 32, tid);

    const bf16* sa = sA[cur] + wib * 16 * KP;
    v16bf a  = load_frag(sa, KP);
    v16bf b0 = load_frag(sB[cur] +  0 * KP, KP);
    v16bf b1 = load_frag(sB[cur] + 16 * KP, KP);
    v16bf b2 = load_frag(sB[cur] + 32 * KP, KP);
    v16bf b3 = load_frag(sB[cur] + 48 * KP, KP);
    acc[0] = wmma_bf16(a, b0, acc[0]);
    acc[1] = wmma_bf16(a, b1, acc[1]);
    acc[2] = wmma_bf16(a, b2, acc[2]);
    acc[3] = wmma_bf16(a, b3, acc[3]);

    wait_async_all();
    __syncthreads();
  }
}

// ---------------------------------------------------------------------------
// QKV projection with scatter epilogue into Q[B,H,T,HD] (x 1/sqrt(HD)),
// K[B,H,T,HD], VT[B,H,HD,T].
// ---------------------------------------------------------------------------
__global__ __launch_bounds__(256) void qkv_gemm(
    const bf16* __restrict__ Xb, const bf16* __restrict__ WaT,
    bf16* __restrict__ Q, bf16* __restrict__ K, bf16* __restrict__ VT) {
  const int nBlocksN = N3C / 64;                 // 48
  int bm = blockIdx.x / nBlocksN;
  int bn = blockIdx.x - bm * nBlocksN;
  int m0 = bm * 128, n0 = bn * 64;

  v8f acc[4] = {};
  gemm_core(Xb + (size_t)m0 * Cn, WaT + (size_t)n0 * Cn, acc);

  int lane  = threadIdx.x & 31;
  int wib   = threadIdx.x >> 5;
  int mbase = m0 + wib * 16;
  int mhalf = (lane >> 4) << 3;
  int nn    = lane & 15;
#pragma unroll
  for (int j = 0; j < 4; ++j) {
    int c     = n0 + j * 16 + nn;     // column in [0,3C)
    int which = c >> 10;              // 0=q 1=k 2=v
    int cc    = c & (Cn - 1);
    int h     = cc >> 6;
    int d     = cc & (HDn - 1);
#pragma unroll
    for (int r = 0; r < 8; ++r) {
      int m = mbase + mhalf + r;      // row in [0, B*T)
      int b = m >> 11;
      int t = m & (Tn - 1);
      float val = acc[j][r];
      size_t bhh = (size_t)(b * Hn + h);
      if (which == 0)
        Q[(bhh * Tn + t) * HDn + d] = (bf16)(val * 0.125f);   // 1/sqrt(64)
      else if (which == 1)
        K[(bhh * Tn + t) * HDn + d] = (bf16)val;
      else
        VT[(bhh * HDn + d) * Tn + t] = (bf16)val;
    }
  }
}

// ---------------------------------------------------------------------------
// Fused causal flash attention. One wave per (b, h, 16-query tile).
// Online softmax state stays register-resident, aligned with the WMMA C/D row
// layout (row lives in one VGPR across a 16-lane half -> shfl_xor reductions).
// P converts C-layout -> A-layout through a per-wave LDS tile.
// ---------------------------------------------------------------------------
__global__ __launch_bounds__(256) void attn_fused(
    const bf16* __restrict__ Q, const bf16* __restrict__ K,
    const bf16* __restrict__ VT, bf16* __restrict__ Y) {
  __shared__ bf16 pbuf[8][16 * KP];
  int lane = threadIdx.x & 31;
  int wib  = threadIdx.x >> 5;
  int wave = blockIdx.x * 8 + wib;
  int qt   = wave & (Tn / 16 - 1);    // 128 q-tiles
  int bh   = wave >> 7;               // b*H + h
  int q0   = qt * 16;

  const bf16* Qp = Q  + (size_t)bh * Tn * HDn + (size_t)q0 * HDn;
  const bf16* Kp = K  + (size_t)bh * Tn * HDn;
  const bf16* Vp = VT + (size_t)bh * HDn * Tn;

  v16bf qa0 = load_frag(Qp, HDn);        // Q resident for whole key loop
  v16bf qa1 = load_frag(Qp + 32, HDn);

  v8f o0 = {}, o1 = {}, o2 = {}, o3 = {};
  v8f mrun, lrun;
#pragma unroll
  for (int r = 0; r < 8; ++r) { mrun[r] = -__builtin_inff(); lrun[r] = 0.0f; }

  int mhalf = (lane >> 4) << 3;
  int nn    = lane & 15;
  bf16* ps  = &pbuf[wib][0];

  for (int k0 = 0; k0 <= q0; k0 += 32) {      // causal truncation
    // ---- S = Q.K^T for 32 keys (4 WMMAs) ----
    v8f s0 = {}, s1 = {};
    {
      v16bf kb0 = load_frag(Kp + (size_t)k0 * HDn,      HDn);
      v16bf kb1 = load_frag(Kp + (size_t)k0 * HDn + 32, HDn);
      s0 = wmma_bf16(qa0, kb0, s0);
      s0 = wmma_bf16(qa1, kb1, s0);
      v16bf kb2 = load_frag(Kp + (size_t)(k0 + 16) * HDn,      HDn);
      v16bf kb3 = load_frag(Kp + (size_t)(k0 + 16) * HDn + 32, HDn);
      s1 = wmma_bf16(qa0, kb2, s1);
      s1 = wmma_bf16(qa1, kb3, s1);
    }
    // ---- issue V fragment loads now; they fly during the softmax VALU ----
    v16bf vb0 = load_frag(Vp + (size_t) 0 * Tn + k0, Tn);
    v16bf vb1 = load_frag(Vp + (size_t)16 * Tn + k0, Tn);
    v16bf vb2 = load_frag(Vp + (size_t)32 * Tn + k0, Tn);
    v16bf vb3 = load_frag(Vp + (size_t)48 * Tn + k0, Tn);

    // causal mask
#pragma unroll
    for (int r = 0; r < 8; ++r) {
      int qrow = q0 + mhalf + r;
      if (k0 + nn > qrow)      s0[r] = -__builtin_inff();
      if (k0 + 16 + nn > qrow) s1[r] = -__builtin_inff();
    }
    // row max across the 16-lane half
    v8f mt;
#pragma unroll
    for (int r = 0; r < 8; ++r) mt[r] = fmaxf(s0[r], s1[r]);
#pragma unroll
    for (int off = 8; off >= 1; off >>= 1)
#pragma unroll
      for (int r = 0; r < 8; ++r) mt[r] = fmaxf(mt[r], __shfl_xor(mt[r], off, 32));

    v8f mnew, alpha;
#pragma unroll
    for (int r = 0; r < 8; ++r) {
      mnew[r]  = fmaxf(mrun[r], mt[r]);
      alpha[r] = __expf(mrun[r] - mnew[r]);   // 0 on first chunk (mrun=-inf)
    }
    // P = exp(S - mnew): stage to LDS for layout conversion; rowsum on the fly
    v8f psum;
#pragma unroll
    for (int r = 0; r < 8; ++r) {
      float p0 = __expf(s0[r] - mnew[r]);
      float p1 = __expf(s1[r] - mnew[r]);
      ps[(mhalf + r) * KP + nn]      = (bf16)p0;
      ps[(mhalf + r) * KP + 16 + nn] = (bf16)p1;
      psum[r] = p0 + p1;
    }
#pragma unroll
    for (int off = 8; off >= 1; off >>= 1)
#pragma unroll
      for (int r = 0; r < 8; ++r) psum[r] += __shfl_xor(psum[r], off, 32);
#pragma unroll
    for (int r = 0; r < 8; ++r) {
      lrun[r] = lrun[r] * alpha[r] + psum[r];
      mrun[r] = mnew[r];
      o0[r] *= alpha[r]; o1[r] *= alpha[r]; o2[r] *= alpha[r]; o3[r] *= alpha[r];
    }
    // wave-local LDS visibility before re-reading P in A-layout
    asm volatile("s_wait_dscnt 0x0" ::: "memory");
    v16bf pa = load_frag(ps, KP);
    o0 = wmma_bf16(pa, vb0, o0);
    o1 = wmma_bf16(pa, vb1, o1);
    o2 = wmma_bf16(pa, vb2, o2);
    o3 = wmma_bf16(pa, vb3, o3);
    asm volatile("s_wait_dscnt 0x0" ::: "memory");  // P reads before overwrite
  }

  // finalize: O / l, write into Y[B*T, C] (bf16) for the projection GEMM
  int b = bh >> 4;
  int h = bh & (Hn - 1);
  size_t rowbase = ((size_t)b * Tn + q0 + mhalf) * Cn + (size_t)h * HDn + nn;
#pragma unroll
  for (int r = 0; r < 8; ++r) {
    float inv = 1.0f / lrun[r];
    Y[rowbase + (size_t)r * Cn +  0] = (bf16)(o0[r] * inv);
    Y[rowbase + (size_t)r * Cn + 16] = (bf16)(o1[r] * inv);
    Y[rowbase + (size_t)r * Cn + 32] = (bf16)(o2[r] * inv);
    Y[rowbase + (size_t)r * Cn + 48] = (bf16)(o3[r] * inv);
  }
}

// ---------------------------------------------------------------------------
// Output projection: out[M,C] (f32) = Y[M,C] (bf16) @ Wp (given as WpT[C,C])
// ---------------------------------------------------------------------------
__global__ __launch_bounds__(256) void proj_gemm(
    const bf16* __restrict__ Yb, const bf16* __restrict__ WpT,
    float* __restrict__ Out) {
  const int nBlocksN = Cn / 64;                  // 16
  int bm = blockIdx.x / nBlocksN;
  int bn = blockIdx.x - bm * nBlocksN;
  int m0 = bm * 128, n0 = bn * 64;

  v8f acc[4] = {};
  gemm_core(Yb + (size_t)m0 * Cn, WpT + (size_t)n0 * Cn, acc);

  int lane  = threadIdx.x & 31;
  int wib   = threadIdx.x >> 5;
  int mbase = m0 + wib * 16;
  int mhalf = (lane >> 4) << 3;
  int nn    = lane & 15;
#pragma unroll
  for (int j = 0; j < 4; ++j) {
#pragma unroll
    for (int r = 0; r < 8; ++r) {
      Out[(size_t)(mbase + mhalf + r) * Cn + n0 + j * 16 + nn] = acc[j][r];
    }
  }
}

// ---------------------------------------------------------------------------
extern "C" void kernel_launch(void* const* d_in, const int* in_sizes, int n_in,
                              void* d_out, int out_size, void* d_ws, size_t ws_size,
                              hipStream_t stream) {
  (void)in_sizes; (void)n_in; (void)out_size; (void)ws_size;
  const float* x      = (const float*)d_in[0];
  const float* w_attn = (const float*)d_in[1];
  const float* w_proj = (const float*)d_in[2];
  float* out = (float*)d_out;

  char* w = (char*)d_ws;
  size_t o = 0;
  bf16* Xb  = (bf16*)(w + o); o += (size_t)Mrows * Cn * 2;   // 16 MB
  bf16* WaT = (bf16*)(w + o); o += (size_t)N3C * Cn * 2;     //  6 MB
  bf16* WpT = (bf16*)(w + o); o += (size_t)Cn * Cn * 2;      //  2 MB
  bf16* Qb  = (bf16*)(w + o); o += (size_t)Mrows * Cn * 2;   // 16 MB
  bf16* Kb  = (bf16*)(w + o); o += (size_t)Mrows * Cn * 2;   // 16 MB
  bf16* VTb = (bf16*)(w + o); o += (size_t)Mrows * Cn * 2;   // 16 MB
  bf16* Yb  = (bf16*)(w + o); o += (size_t)Mrows * Cn * 2;   // 16 MB  (88 MB total)

  int nx = Mrows * Cn;
  cvt_f32_bf16 <<<nx / 256, 256, 0, stream>>>(x, Xb, nx);
  transpose_cvt<<<(N3C * Cn) / 256, 256, 0, stream>>>(w_attn, WaT, Cn, N3C);
  transpose_cvt<<<(Cn  * Cn) / 256, 256, 0, stream>>>(w_proj, WpT, Cn, Cn);

  qkv_gemm <<<(Mrows / 128) * (N3C / 64), 256, 0, stream>>>(Xb, WaT, Qb, Kb, VTb);
  attn_fused<<<(Bn * Hn * (Tn / 16)) / 8,  256, 0, stream>>>(Qb, Kb, VTb, Yb);
  proj_gemm <<<(Mrows / 128) * (Cn / 64),  256, 0, stream>>>(Yb, WpT, out);
}
